// MLA_19258633356205
// MI455X (gfx1250) — compile-verified, matrix operand order checked by reading, not run on hardware
//
#include <hip/hip_runtime.h>
#include <cstdint>
#include <cstddef>

// ---------------- problem constants ----------------
#define DIM_   2048
#define NH_    16
#define QLR_   1536
#define KVLR_  512
#define NOPE_  128
#define ROPE_  64
#define VD_    128
#define B_     2
#define S_     2048
#define QKHD_  192          // NOPE + ROPE
#define MROWS_ 4096         // B * S
#define SCALE_ 0.07216878364870323f   // (192)^-0.5
#define EPS_   1.1920929e-7f

typedef __attribute__((ext_vector_type(16))) __bf16 v16bf;
typedef __attribute__((ext_vector_type(8)))  float  v8f;

static_assert(MROWS_ % 128 == 0, "");
static_assert(QKHD_ % 32 == 0, "");

// ---------------- fragment helpers ----------------

__device__ __forceinline__ v8f v8zero() {
    v8f z = {0.f, 0.f, 0.f, 0.f, 0.f, 0.f, 0.f, 0.f};
    return z;
}

// Load a 16x32 bf16 fragment from a row-major [16+, ld] matrix.
// Matches CDNA5 16-bit A layout (and symmetric B layout when the B matrix
// is stored transposed, i.e. N x K row-major):
//   lanes 0-15 : row = lane,     K = 0..7 (V0-3), 16..23 (V4-7)
//   lanes 16-31: row = lane-16,  K = 8..15,       24..31
__device__ __forceinline__ v16bf load_frag_nk(const __bf16* p, int ld) {
    int lane = threadIdx.x & 31;
    const __bf16* r0 = p + (size_t)(lane & 15) * ld + ((lane >> 4) << 3);
    union { uint4 u[2]; v16bf f; } u;
    u.u[0] = *(const uint4*)r0;          // K: hk .. hk+7
    u.u[1] = *(const uint4*)(r0 + 16);   // K: 16+hk .. 16+hk+7
    return u.f;
}

__device__ __forceinline__ v8f wmma_bf16(v16bf a, v16bf b, v8f c) {
    return __builtin_amdgcn_wmma_f32_16x16x32_bf16(
        false, a, false, b, (short)0, c, false, false);
}

// ---------------- elementwise cast ----------------

__global__ void cast_bf16_kernel(__bf16* __restrict__ dst,
                                 const float* __restrict__ src, size_t n) {
    size_t i = (size_t)blockIdx.x * blockDim.x + threadIdx.x;
    if (i < n) dst[i] = (__bf16)src[i];
}

// ---------------- generic WMMA GEMM: C = A * W^T + bias ----------------
// A: [M,K] bf16 row-major, W: [N,K] bf16 row-major, C: [M,N] f32.
// Block: 128 threads (4 waves) -> 128x64 tile; each wave owns 32x64
// (2 M-frags x 4 N-frags => 8 WMMAs per 6 fragment loads per K-step).
__global__ void gemm_bias_bf16(float* __restrict__ C,
                               const __bf16* __restrict__ A,
                               const __bf16* __restrict__ W,
                               const float* __restrict__ bias,
                               int M, int N, int K) {
    int wave = threadIdx.x >> 5;
    int m0 = blockIdx.x * 128 + wave * 32;
    int n0 = blockIdx.y * 64;

    v8f acc[2][4];
    #pragma unroll
    for (int i = 0; i < 2; ++i)
        #pragma unroll
        for (int j = 0; j < 4; ++j) acc[i][j] = v8zero();

    for (int k = 0; k < K; k += 32) {
        v16bf a0 = load_frag_nk(A + (size_t)m0 * K + k, K);
        v16bf a1 = load_frag_nk(A + (size_t)(m0 + 16) * K + k, K);
        v16bf b[4];
        #pragma unroll
        for (int j = 0; j < 4; ++j)
            b[j] = load_frag_nk(W + (size_t)(n0 + 16 * j) * K + k, K);
        #pragma unroll
        for (int j = 0; j < 4; ++j) acc[0][j] = wmma_bf16(a0, b[j], acc[0][j]);
        #pragma unroll
        for (int j = 0; j < 4; ++j) acc[1][j] = wmma_bf16(a1, b[j], acc[1][j]);
    }

    int lane = threadIdx.x & 31;
    int col  = lane & 15;
    int rb   = (lane >> 4) << 3;      // row base within 16x16 tile
    for (int i = 0; i < 2; ++i) {
        for (int j = 0; j < 4; ++j) {
            float bv = bias[n0 + 16 * j + col];
            float* Cp = C + (size_t)(m0 + 16 * i) * N + n0 + 16 * j;
            #pragma unroll
            for (int r = 0; r < 8; ++r)
                Cp[(size_t)(rb + r) * N + col] = acc[i][j][r] + bv;
        }
    }
}

// ---------------- RMSNorm: f32 in -> bf16 out ----------------
__global__ void rmsnorm_bf16(__bf16* __restrict__ dst,
                             const float* __restrict__ src,
                             const float* __restrict__ w,
                             int width, int src_stride) {
    int row = blockIdx.x;
    const float* s = src + (size_t)row * src_stride;
    __bf16* d = dst + (size_t)row * width;

    float ss = 0.f;
    for (int i = threadIdx.x; i < width; i += blockDim.x) {
        float v = s[i];
        ss += v * v;
    }
    __shared__ float red[8];
    for (int off = 16; off; off >>= 1) ss += __shfl_xor(ss, off, 32);
    int wave = threadIdx.x >> 5, lane = threadIdx.x & 31;
    if (lane == 0) red[wave] = ss;
    __syncthreads();
    if (wave == 0) {
        float v = (lane < (blockDim.x >> 5)) ? red[lane] : 0.f;
        for (int off = 16; off; off >>= 1) v += __shfl_xor(v, off, 32);
        if (lane == 0) red[0] = v;
    }
    __syncthreads();
    float inv = rsqrtf(red[0] / (float)width + EPS_);
    for (int i = threadIdx.x; i < width; i += blockDim.x)
        d[i] = (__bf16)(s[i] * inv * w[i]);
}

// ---------------- prep: q (RoPE on last 64) -> Qh[b,h,s,192] bf16 ----------
__global__ void prep_q(__bf16* __restrict__ Qh, const float* __restrict__ q,
                       const float* __restrict__ cosp,
                       const float* __restrict__ sinp) {
    int idx = blockIdx.x;                 // b*S*NH + s*NH + h
    int h = idx % NH_;
    int s = (idx / NH_) % S_;
    int b = idx / (NH_ * S_);
    const float* src = q + ((size_t)(b * S_ + s) * NH_ + h) * QKHD_;
    __bf16* dst = Qh + (((size_t)(b * NH_ + h) * S_) + s) * QKHD_;
    int d = threadIdx.x;                  // 0..191
    float v;
    if (d < NOPE_) {
        v = src[d];
    } else {
        int j = d - NOPE_;
        int pair = j >> 1;
        float c  = cosp[s * (ROPE_ / 2) + pair];
        float sn = sinp[s * (ROPE_ / 2) + pair];
        float xe = src[NOPE_ + 2 * pair];
        float xo = src[NOPE_ + 2 * pair + 1];
        v = (j & 1) ? (xe * sn + xo * c) : (xe * c - xo * sn);
    }
    dst[d] = (__bf16)v;
}

// ---------------- prep: ckv -> Kh nope part + Vt (transposed V) ------------
__global__ void prep_kv(__bf16* __restrict__ Kh, __bf16* __restrict__ Vt,
                        const float* __restrict__ ckv) {
    int idx = blockIdx.x;                 // b*S*NH + s*NH + h
    int h = idx % NH_;
    int s = (idx / NH_) % S_;
    int b = idx / (NH_ * S_);
    const float* src = ckv + ((size_t)(b * S_ + s) * NH_ + h) * (NOPE_ + VD_);
    int d = threadIdx.x;                  // 0..255
    if (d < NOPE_) {
        Kh[(((size_t)(b * NH_ + h) * S_) + s) * QKHD_ + d] = (__bf16)src[d];
    } else {
        int vd = d - NOPE_;
        Vt[(((size_t)(b * NH_ + h) * VD_) + vd) * S_ + s] = (__bf16)src[d];
    }
}

// ---------------- prep: k_rope (RoPE, broadcast to all heads) --------------
__global__ void prep_krope(__bf16* __restrict__ Kh, const float* __restrict__ kv,
                           const float* __restrict__ cosp,
                           const float* __restrict__ sinp) {
    int s = blockIdx.x % S_;
    int b = blockIdx.x / S_;
    const float* src = kv + (size_t)(b * S_ + s) * (KVLR_ + ROPE_) + KVLR_;
    int j = threadIdx.x;                  // 0..63
    int pair = j >> 1;
    float c  = cosp[s * (ROPE_ / 2) + pair];
    float sn = sinp[s * (ROPE_ / 2) + pair];
    float xe = src[2 * pair];
    float xo = src[2 * pair + 1];
    float v = (j & 1) ? (xe * sn + xo * c) : (xe * c - xo * sn);
    __bf16 bv = (__bf16)v;
    for (int h = 0; h < NH_; ++h)
        Kh[(((size_t)(b * NH_ + h) * S_) + s) * QKHD_ + NOPE_ + j] = bv;
}

// ---------------- causal flash attention, one wave = 16 query rows ---------
// grid: (S/64, NH, B), block: 128 threads (4 waves).
// 64-key blocks: 24 QK WMMAs + 16 PV WMMAs per block, softmax reductions
// amortized over 64 keys.
__global__ void mla_attn(__bf16* __restrict__ O,
                         const __bf16* __restrict__ Qh,
                         const __bf16* __restrict__ Kh,
                         const __bf16* __restrict__ Vt) {
    __shared__ __align__(16) __bf16 plds[4][16 * 64];   // per-wave P tile

    int wave = threadIdx.x >> 5;
    int lane = threadIdx.x & 31;
    int col  = lane & 15;                 // key/vd column within 16x16 tile
    int rb   = (lane >> 4) << 3;          // query-row base within tile

    int b = blockIdx.z, h = blockIdx.y;
    int q0 = blockIdx.x * 64 + wave * 16;

    const __bf16* Qp = Qh + (((size_t)(b * NH_ + h) * S_) + q0) * QKHD_;
    const __bf16* Kp = Kh + ((size_t)(b * NH_ + h) * S_) * QKHD_;
    const __bf16* Vp = Vt + ((size_t)(b * NH_ + h) * VD_) * S_;

    // Q fragments resident in registers: 192 = 6 * 32
    v16bf aq[6];
    #pragma unroll
    for (int t = 0; t < 6; ++t) aq[t] = load_frag_nk(Qp + 32 * t, QKHD_);

    v8f o[8];
    #pragma unroll
    for (int j = 0; j < 8; ++j) o[j] = v8zero();
    float m_i[8], l_i[8];
    #pragma unroll
    for (int r = 0; r < 8; ++r) { m_i[r] = -INFINITY; l_i[r] = 0.f; }

    __bf16* pl = plds[wave];
    int kend = q0 + 16;                   // causal: keys < kend
    for (int kb = 0; kb < kend; kb += 64) {
        // scores: 16 rows x 64 keys
        v8f sc[4];
        #pragma unroll
        for (int j = 0; j < 4; ++j) sc[j] = v8zero();
        #pragma unroll
        for (int t = 0; t < 6; ++t) {
            #pragma unroll
            for (int j = 0; j < 4; ++j) {
                v16bf bk = load_frag_nk(
                    Kp + (size_t)(kb + 16 * j) * QKHD_ + 32 * t, QKHD_);
                sc[j] = wmma_bf16(aq[t], bk, sc[j]);
            }
        }
        // online softmax, per 8 rows owned by this lane's half
        #pragma unroll
        for (int r = 0; r < 8; ++r) {
            int mrow = q0 + rb + r;
            float s[4];
            #pragma unroll
            for (int j = 0; j < 4; ++j)
                s[j] = (kb + 16 * j + col <= mrow) ? sc[j][r] * SCALE_
                                                   : -INFINITY;
            float mx = fmaxf(fmaxf(s[0], s[1]), fmaxf(s[2], s[3]));
            for (int off = 1; off < 16; off <<= 1)
                mx = fmaxf(mx, __shfl_xor(mx, off, 32));
            float mn = fmaxf(m_i[r], mx);
            float p[4], rs = 0.f;
            #pragma unroll
            for (int j = 0; j < 4; ++j) {
                p[j] = __expf(s[j] - mn);
                rs += p[j];
            }
            for (int off = 1; off < 16; off <<= 1)
                rs += __shfl_xor(rs, off, 32);
            float corr = __expf(m_i[r] - mn);
            l_i[r] = l_i[r] * corr + rs;
            m_i[r] = mn;
            #pragma unroll
            for (int j = 0; j < 8; ++j) o[j][r] *= corr;
            #pragma unroll
            for (int j = 0; j < 4; ++j)
                pl[(rb + r) * 64 + 16 * j + col] = (__bf16)p[j];
        }
        // P (C-layout) -> A-layout via per-wave LDS (LDS ops in-order per wave)
        v16bf pa0 = load_frag_nk(pl, 64);        // keys kb   .. kb+31
        v16bf pa1 = load_frag_nk(pl + 32, 64);   // keys kb+32.. kb+63
        #pragma unroll
        for (int j = 0; j < 8; ++j) {
            v16bf bv0 = load_frag_nk(Vp + (size_t)(16 * j) * S_ + kb, S_);
            v16bf bv1 = load_frag_nk(Vp + (size_t)(16 * j) * S_ + kb + 32, S_);
            o[j] = wmma_bf16(pa0, bv0, o[j]);
            o[j] = wmma_bf16(pa1, bv1, o[j]);
        }
    }

    // normalize + store bf16 to O[b, s, h*VD + d]
    __bf16* Op = O + ((size_t)(b * S_) + q0) * (NH_ * VD_) + h * VD_;
    #pragma unroll
    for (int r = 0; r < 8; ++r) {
        float inv = 1.0f / l_i[r];
        #pragma unroll
        for (int j = 0; j < 8; ++j)
            Op[(size_t)(rb + r) * (NH_ * VD_) + 16 * j + col] =
                (__bf16)(o[j][r] * inv);
    }
}

// ---------------- host-side orchestration ----------------

static inline void launch_cast(__bf16* dst, const float* src, size_t n,
                               hipStream_t st) {
    int blocks = (int)((n + 255) / 256);
    cast_bf16_kernel<<<blocks, 256, 0, st>>>(dst, src, n);
}

extern "C" void kernel_launch(void* const* d_in, const int* in_sizes, int n_in,
                              void* d_out, int out_size, void* d_ws,
                              size_t ws_size, hipStream_t stream) {
    (void)in_sizes; (void)n_in; (void)out_size; (void)ws_size;

    const float* x     = (const float*)d_in[0];
    // d_in[1] = start_pos (always 0), d_in[4] = mask (causal, rebuilt inline)
    const float* cosp  = (const float*)d_in[2];
    const float* sinp  = (const float*)d_in[3];
    const float* wqaw  = (const float*)d_in[5];
    const float* wqab  = (const float*)d_in[6];
    const float* wqbw  = (const float*)d_in[7];
    const float* wqbb  = (const float*)d_in[8];
    const float* wkvaw = (const float*)d_in[9];
    const float* wkvab = (const float*)d_in[10];
    const float* wkvbw = (const float*)d_in[11];
    const float* wkvbb = (const float*)d_in[12];
    const float* wow   = (const float*)d_in[13];
    const float* wob   = (const float*)d_in[14];
    const float* qnw   = (const float*)d_in[15];
    const float* knw   = (const float*)d_in[16];
    float* out = (float*)d_out;

    char* base = (char*)d_ws;
    size_t off = 0;
    auto alloc = [&](size_t bytes) -> char* {
        char* p = base + off;
        off = (off + bytes + 255) & ~(size_t)255;
        return p;
    };

    __bf16* xb   = (__bf16*)alloc((size_t)MROWS_ * DIM_ * 2);
    __bf16* wqa  = (__bf16*)alloc((size_t)QLR_ * DIM_ * 2);
    __bf16* wqb  = (__bf16*)alloc((size_t)(NH_ * QKHD_) * QLR_ * 2);
    __bf16* wkva = (__bf16*)alloc((size_t)(KVLR_ + ROPE_) * DIM_ * 2);
    __bf16* wkvb = (__bf16*)alloc((size_t)(NH_ * (NOPE_ + VD_)) * KVLR_ * 2);
    __bf16* wo   = (__bf16*)alloc((size_t)DIM_ * (NH_ * VD_) * 2);
    float*  qa   = (float*) alloc((size_t)MROWS_ * QLR_ * 4);
    __bf16* qn   = (__bf16*)alloc((size_t)MROWS_ * QLR_ * 2);
    float*  qf   = (float*) alloc((size_t)MROWS_ * (NH_ * QKHD_) * 4);
    float*  kvf  = (float*) alloc((size_t)MROWS_ * (KVLR_ + ROPE_) * 4);
    __bf16* ckvn = (__bf16*)alloc((size_t)MROWS_ * KVLR_ * 2);
    float*  ckv  = (float*) alloc((size_t)MROWS_ * (NH_ * (NOPE_ + VD_)) * 4);
    __bf16* Qh   = (__bf16*)alloc((size_t)B_ * NH_ * S_ * QKHD_ * 2);
    __bf16* Kh   = (__bf16*)alloc((size_t)B_ * NH_ * S_ * QKHD_ * 2);
    __bf16* Vt   = (__bf16*)alloc((size_t)B_ * NH_ * VD_ * S_ * 2);
    __bf16* Ob   = (__bf16*)alloc((size_t)MROWS_ * (NH_ * VD_) * 2);

    // 1) bf16 casts (x + weights)
    launch_cast(xb,   x,     (size_t)MROWS_ * DIM_, stream);
    launch_cast(wqa,  wqaw,  (size_t)QLR_ * DIM_, stream);
    launch_cast(wqb,  wqbw,  (size_t)(NH_ * QKHD_) * QLR_, stream);
    launch_cast(wkva, wkvaw, (size_t)(KVLR_ + ROPE_) * DIM_, stream);
    launch_cast(wkvb, wkvbw, (size_t)(NH_ * (NOPE_ + VD_)) * KVLR_, stream);
    launch_cast(wo,   wow,   (size_t)DIM_ * (NH_ * VD_), stream);

    // 2) q_a = x @ wq_a^T + b   [4096 x 1536]
    {
        dim3 g(MROWS_ / 128, QLR_ / 64);
        gemm_bias_bf16<<<g, 128, 0, stream>>>(qa, xb, wqa, wqab,
                                              MROWS_, QLR_, DIM_);
    }
    // 3) rmsnorm(q_a) -> qn bf16
    rmsnorm_bf16<<<MROWS_, 256, 0, stream>>>(qn, qa, qnw, QLR_, QLR_);

    // 4) q = qn @ wq_b^T + b    [4096 x 3072]
    {
        dim3 g(MROWS_ / 128, (NH_ * QKHD_) / 64);
        gemm_bias_bf16<<<g, 128, 0, stream>>>(qf, qn, wqb, wqbb,
                                              MROWS_, NH_ * QKHD_, QLR_);
    }
    // 5) kv = x @ wkv_a^T + b   [4096 x 576]
    {
        dim3 g(MROWS_ / 128, (KVLR_ + ROPE_) / 64);
        gemm_bias_bf16<<<g, 128, 0, stream>>>(kvf, xb, wkva, wkvab,
                                              MROWS_, KVLR_ + ROPE_, DIM_);
    }
    // 6) rmsnorm(c_kv) -> ckvn bf16 (first 512 of each 576-wide row)
    rmsnorm_bf16<<<MROWS_, 256, 0, stream>>>(ckvn, kvf, knw, KVLR_,
                                             KVLR_ + ROPE_);
    // 7) ckv = ckvn @ wkv_b^T + b  [4096 x 4096]
    {
        dim3 g(MROWS_ / 128, (NH_ * (NOPE_ + VD_)) / 64);
        gemm_bias_bf16<<<g, 128, 0, stream>>>(ckv, ckvn, wkvb, wkvbb,
                                              MROWS_, NH_ * (NOPE_ + VD_),
                                              KVLR_);
    }
    // 8) layout/RoPE prep
    prep_q<<<B_ * S_ * NH_, QKHD_, 0, stream>>>(Qh, qf, cosp, sinp);
    prep_kv<<<B_ * S_ * NH_, NOPE_ + VD_, 0, stream>>>(Kh, Vt, ckv);
    prep_krope<<<B_ * S_, ROPE_, 0, stream>>>(Kh, kvf, cosp, sinp);

    // 9) causal attention -> Ob bf16 [4096 x 2048]
    {
        dim3 g(S_ / 64, NH_, B_);
        mla_attn<<<g, 128, 0, stream>>>(Ob, Qh, Kh, Vt);
    }
    // 10) out = Ob @ wo^T + b   [4096 x 2048] f32
    {
        dim3 g(MROWS_ / 128, DIM_ / 64);
        gemm_bias_bf16<<<g, 128, 0, stream>>>(out, Ob, wo, wob,
                                              MROWS_, DIM_, NH_ * VD_);
    }
}
